// P4DConv_42777874268412
// MI455X (gfx1250) — matrix operand order repaired and checked by reading, not compiled, not run on hardware
//
#include <hip/hip_runtime.h>
#include <hip/hip_bf16.h>

#define B_   4
#define T_   8
#define N_   4096
#define CIN  64
#define C0_  64
#define C1_  128
#define M_   1024
#define KNN  32
#define R2_  0.04f
#define NWAVES 4
#define CONV_BLOCKS 2048   // persistent grid: 8192 waves, 12 anchor-tasks each

typedef __attribute__((ext_vector_type(16))) _Float16 v16h;
typedef __attribute__((ext_vector_type(8)))  float    v8f;

__device__ __forceinline__ v8f wmma16(v16h a, v16h b, v8f c) {
  // D = A(16x32 f16) * B(32x16 f16) + C(16x16 f32)
  return __builtin_amdgcn_wmma_f32_16x16x32_f16(
      /*neg_a=*/false, a, /*neg_b=*/false, b,
      /*c_mod=*/(short)0, c, /*reuse_a=*/false, /*reuse_b=*/false);
}

// Load a WMMA operand fragment from an LDS tile stored row-major [row][64] f16.
// PAT=0: A-operand layout (16-bit A 16x32, ISA 7.12.2) -> two contiguous 8-half runs
// PAT=1: B-operand layout, same k order as PAT=0 -> two contiguous 8-half runs
template <int PAT>
__device__ __forceinline__ v16h load_frag(const _Float16* tile, int row, int kbase, int lane) {
  const _Float16* rp = tile + row * 64;
  const int hi = lane >> 4;
  v16h f;
#pragma unroll
  for (int p = 0; p < 8; ++p) {
    int k;
    if (PAT == 0) k = kbase + ((p & 3) << 1) + ((p >> 2) << 4) + (hi << 3);
    else          k = kbase + ((p & 3) << 1) + ((p >> 2) << 4) + (hi << 3);
    f[2 * p]     = rp[k];
    f[2 * p + 1] = rp[k + 1];
  }
  return f;
}

__device__ __forceinline__ float xor16_max(float v) {
  int sw = __builtin_amdgcn_ds_swizzle(__float_as_int(v), 0x401F);  // xor=16
  return fmaxf(v, __int_as_float(sw));
}

// argmax combine matching jnp.argmax first-occurrence semantics
__device__ __forceinline__ void amx_combine(float& v, int& i, float v2, int i2) {
  if (v2 > v || (v2 == v && i2 < i)) { v = v2; i = i2; }
}
template <int OFF>
__device__ __forceinline__ void amx_swz(float& v, int& i) {
  float v2 = __int_as_float(__builtin_amdgcn_ds_swizzle(__float_as_int(v), OFF));
  int   i2 = __builtin_amdgcn_ds_swizzle(i, OFF);
  amx_combine(v, i, v2, i2);
}

__global__ void zero_kernel(float* __restrict__ p, int n) {
  int i = blockIdx.x * blockDim.x + threadIdx.x;
  if (i < n) p[i] = 0.f;
}

// ---------------- FPS: one block per (b, s); swizzle-based argmax ----------------
__global__ __launch_bounds__(256) void fps_kernel(const float* __restrict__ xyzs,
                                                  float* __restrict__ out) {
  const int s = blockIdx.x & 7;
  const int b = blockIdx.x >> 3;
  const float* P = xyzs + ((size_t)(b * T_ + s)) * N_ * 3;
  float* anchors = out + ((size_t)(b * 8 + s)) * M_ * 3;
  const int tid  = threadIdx.x;
  const int lane = tid & 31;
  const int wv8  = tid >> 5;

  __shared__ float redV[8];
  __shared__ int   redI[8];
  __shared__ int   sLast;

  float px[16], py[16], pz[16], dist[16];
#pragma unroll
  for (int k = 0; k < 16; ++k) {
    const int j = tid * 16 + k;
    px[k] = P[3 * j]; py[k] = P[3 * j + 1]; pz[k] = P[3 * j + 2];
    dist[k] = 1e10f;
  }
  if (tid == 0) { anchors[0] = P[0]; anchors[1] = P[1]; anchors[2] = P[2]; }

  int last = 0;
  for (int it = 1; it < M_; ++it) {
    const float lx = P[3 * last], ly = P[3 * last + 1], lz = P[3 * last + 2];
    float bv = -1.f; int bi = 0;
#pragma unroll
    for (int k = 0; k < 16; ++k) {
      const float dx = px[k] - lx, dy = py[k] - ly, dz = pz[k] - lz;
      const float d = fminf(dist[k], dx * dx + dy * dy + dz * dz);
      dist[k] = d;
      if (d > bv) { bv = d; bi = tid * 16 + k; }   // ascending j keeps first max
    }
    // intra-wave argmax reduction (no barriers)
    amx_swz<0x041F>(bv, bi);   // xor 1
    amx_swz<0x081F>(bv, bi);   // xor 2
    amx_swz<0x101F>(bv, bi);   // xor 4
    amx_swz<0x201F>(bv, bi);   // xor 8
    amx_swz<0x401F>(bv, bi);   // xor 16
    if (lane == 0) { redV[wv8] = bv; redI[wv8] = bi; }
    __syncthreads();
    if (wv8 == 0) {
      float v = redV[lane & 7]; int idx = redI[lane & 7];
      amx_swz<0x041F>(v, idx);
      amx_swz<0x081F>(v, idx);
      amx_swz<0x101F>(v, idx);
      if (lane == 0) {
        sLast = idx;
        anchors[3 * it]     = P[3 * idx];
        anchors[3 * it + 1] = P[3 * idx + 1];
        anchors[3 * it + 2] = P[3 * idx + 2];
      }
    }
    __syncthreads();
    last = sLast;
  }
}

// ---------------- ball query + MLP + max-pool: persistent waves, one anchor/task ----------------
__global__ __launch_bounds__(128) void p4d_conv_kernel(
    const float* __restrict__ xyzs, const float* __restrict__ features,
    const float* __restrict__ Wd, const float* __restrict__ Wf,
    const float* __restrict__ W1, float* __restrict__ out) {
  __shared__ _Float16 sWf[C0_ * 64];      // [o][c]
  __shared__ _Float16 sW1[C1_ * 64];      // [o][c]
  __shared__ float    sWd[C0_ * 4];
  __shared__ _Float16 sGf[NWAVES][KNN * 64];   // gathered features, [point][c]
  __shared__ _Float16 sH [NWAVES][16 * 64];    // hidden tile, [row][c]
  __shared__ float    sDisp[NWAVES][KNN * 4];
  __shared__ int      sIdx[NWAVES][KNN];

  // stage weights once per (persistent) block
  for (int i = threadIdx.x; i < C0_ * 64; i += blockDim.x) sWf[i] = (_Float16)Wf[i];
  for (int i = threadIdx.x; i < C1_ * 64; i += blockDim.x) sW1[i] = (_Float16)W1[i];
  for (int i = threadIdx.x; i < C0_ * 4;  i += blockDim.x) sWd[i] = Wd[i];
  __syncthreads();

  const int lane = threadIdx.x & 31;
  const int wv   = threadIdx.x >> 5;
  const int lrow = lane & 15;
  const int totalTasks = B_ * 8 * 3 * M_;
  const int waveStride = CONV_BLOCKS * NWAVES;

  for (int task = blockIdx.x * NWAVES + wv; task < totalTasks; task += waveStride) {
    int wid = task;
    const int m    = wid & (M_ - 1); wid >>= 10;
    const int irel = wid % 3;        wid /= 3;
    const int s    = wid & 7;
    const int b    = wid >> 3;
    int f = s + irel - 1; f = f < 0 ? 0 : (f > T_ - 1 ? T_ - 1 : f);
    const float toff = (float)(irel - 1);

    const float* anc = out + (((size_t)(b * 8 + s)) * M_ + m) * 3;
    const float ax = anc[0], ay = anc[1], az = anc[2];
    const float* Pf = xyzs + ((size_t)(b * T_ + f)) * N_ * 3;
    const float* Ff = features + ((size_t)(b * T_ + f)) * CIN * N_;

    // ---- ball query: first KNN indices (ascending) with d2 < R^2 ----
    int cnt = 0;
    for (int base = 0; base < N_ && cnt < KNN; base += 32) {
      const int j = base + lane;
      __builtin_prefetch((const void*)(Pf + 3 * (j + 32)), 0, 3);  // next chunk -> global_prefetch
      const float dx = Pf[3 * j] - ax, dy = Pf[3 * j + 1] - ay, dz = Pf[3 * j + 2] - az;
      const bool in = (dx * dx + dy * dy + dz * dz) < R2_;
      const unsigned mask = __builtin_amdgcn_ballot_w32(in);
      if (in) {
        const int pos = cnt + __popc(mask & ((1u << lane) - 1u));
        if (pos < KNN) sIdx[wv][pos] = j;
      }
      cnt += __popc(mask);
    }
    asm volatile("s_wait_dscnt 0" ::: "memory");
    if (cnt > KNN) cnt = KNN;
    int nb = 0;
    if (cnt > 0) nb = (lane < cnt) ? sIdx[wv][lane] : sIdx[wv][0];

    // ---- gather: lane L owns neighbor L ----
    const float gx = Pf[3 * nb], gy = Pf[3 * nb + 1], gz = Pf[3 * nb + 2];
    sDisp[wv][lane * 4 + 0] = gx - ax;
    sDisp[wv][lane * 4 + 1] = gy - ay;
    sDisp[wv][lane * 4 + 2] = gz - az;
    sDisp[wv][lane * 4 + 3] = toff;
    const float* fb = Ff + nb;
#pragma unroll 8
    for (int c = 0; c < CIN; ++c)
      sGf[wv][lane * 64 + c] = (_Float16)fb[(size_t)c * N_];
    asm volatile("s_wait_dscnt 0" ::: "memory");

    float cmax[8];
#pragma unroll
    for (int q = 0; q < 8; ++q) cmax[q] = 0.f;

#pragma unroll
    for (int rt = 0; rt < 2; ++rt) {     // two 16-neighbor row tiles
      v16h A0 = load_frag<0>(sGf[wv], rt * 16 + lrow, 0,  lane);
      v16h A1 = load_frag<0>(sGf[wv], rt * 16 + lrow, 32, lane);

      // stage A: h = relu(gf @ Wf^T) + relu(d4 @ Wd^T)
#pragma unroll
      for (int nt = 0; nt < 4; ++nt) {
        v16h B0 = load_frag<1>(sWf, nt * 16 + lrow, 0,  lane);
        v16h B1 = load_frag<1>(sWf, nt * 16 + lrow, 32, lane);
        v8f acc = {};
        acc = wmma16(A0, B0, acc);
        acc = wmma16(A1, B1, acc);
        const int col = nt * 16 + lrow;
        const float w0 = sWd[col * 4 + 0], w1 = sWd[col * 4 + 1];
        const float w2 = sWd[col * 4 + 2], w3 = sWd[col * 4 + 3];
        const int hi8 = (lane >> 4) << 3;
#pragma unroll
        for (int r = 0; r < 8; ++r) {
          const int row = r + hi8;       // C/D layout: VGPR r -> row r (+8 for hi lanes)
          const float* dp = &sDisp[wv][(rt * 16 + row) * 4];
          const float dot = dp[0] * w0 + dp[1] * w1 + dp[2] * w2 + dp[3] * w3;
          const float hv = fmaxf(acc[r], 0.f) + fmaxf(dot, 0.f);
          sH[wv][row * 64 + col] = (_Float16)hv;
        }
      }
      asm volatile("s_wait_dscnt 0" ::: "memory");

      // stage B: y = relu(h @ W1^T); running max over neighbors
      v16h H0 = load_frag<0>(sH[wv], lrow, 0,  lane);
      v16h H1 = load_frag<0>(sH[wv], lrow, 32, lane);
#pragma unroll
      for (int nt = 0; nt < 8; ++nt) {
        v16h B0 = load_frag<1>(sW1, nt * 16 + lrow, 0,  lane);
        v16h B1 = load_frag<1>(sW1, nt * 16 + lrow, 32, lane);
        v8f acc = {};
        acc = wmma16(H0, B0, acc);
        acc = wmma16(H1, B1, acc);
        float mx = 0.f;                            // relu folded into max(.,0)
#pragma unroll
        for (int r = 0; r < 8; ++r) mx = fmaxf(mx, acc[r]);
        mx = xor16_max(mx);                        // combine row halves (L <-> L^16)
        cmax[nt] = fmaxf(cmax[nt], mx);
      }
      asm volatile("s_wait_dscnt 0" ::: "memory");
    }

    // accumulate over temporal frames: out_feats[b][s][o][m] += max_k
    if (lane < 16) {
      float* of = out + (size_t)B_ * 8 * M_ * 3 + (((size_t)(b * 8 + s)) * C1_) * M_ + m;
#pragma unroll
      for (int nt = 0; nt < 8; ++nt)
        atomicAdd(of + (size_t)(nt * 16 + lane) * M_, cmax[nt]);
    }
  }
}

extern "C" void kernel_launch(void* const* d_in, const int* in_sizes, int n_in,
                              void* d_out, int out_size, void* d_ws, size_t ws_size,
                              hipStream_t stream) {
  const float* xyzs     = (const float*)d_in[0];
  const float* features = (const float*)d_in[1];
  const float* Wd       = (const float*)d_in[2];
  const float* Wf       = (const float*)d_in[3];
  const float* W1       = (const float*)d_in[4];
  float* out = (float*)d_out;

  const int xyzTotal  = B_ * 8 * M_ * 3;
  const int featTotal = B_ * 8 * C1_ * M_;
  zero_kernel<<<(featTotal + 255) / 256, 256, 0, stream>>>(out + xyzTotal, featTotal);
  fps_kernel<<<B_ * 8, 256, 0, stream>>>(xyzs, out);
  p4d_conv_kernel<<<CONV_BLOCKS, 128, 0, stream>>>(xyzs, features, Wd, Wf, W1, out);
}
